// PanoSA_90348932038790
// MI455X (gfx1250) — matrix operand order repaired and checked by reading, not compiled, not run on hardware
//
#include <hip/hip_runtime.h>
#include <hip/hip_bf16.h>
#include <math.h>

typedef __attribute__((ext_vector_type(16))) __bf16 v16bf;
typedef __attribute__((ext_vector_type(8)))  float  v8f;

#define M_   65536           // B*H*W token rows
#define C_   256
#define DFF_ 1024
#define PI_  3.14159265358979323846f

__device__ __forceinline__ unsigned short f2bf(float f){
  unsigned int u = __float_as_uint(f);
  u += 0x7FFFu + ((u >> 16) & 1u);
  return (unsigned short)(u >> 16);
}
__device__ __forceinline__ float bf2f(unsigned short h){
  return __uint_as_float(((unsigned int)h) << 16);
}
// pixel (b,h,w) -> window-token row index
__device__ __forceinline__ int winmap(int b, int h, int w){
  return ((b*128 + (h>>3)*16 + (w>>3))<<6) + ((h&7)<<3) + (w&7);
}
// low 32 bits of a generic pointer to LDS == LDS byte offset
__device__ __forceinline__ unsigned int lds_addr(const void* p){
  return (unsigned int)(unsigned long long)(size_t)p;
}
// CDNA5 async copy: 16 bytes per lane, global -> LDS, tracked by ASYNCcnt
__device__ __forceinline__ void async_copy_b128(unsigned int lds_off, const void* gaddr){
  asm volatile("global_load_async_to_lds_b128 %0, %1, off"
               :: "v"(lds_off), "v"((unsigned long long)(size_t)gaddr)
               : "memory");
}

union FragU { uint4 u[2]; v16bf v; };

// A-matrix 16x32 bf16 fragment from row-major A[.,ldA] (contiguous K order)
__device__ __forceinline__ v16bf load_a_frag(const unsigned short* A, int ldA,
                                             int row0, int k0, int lane){
  int half = lane >> 4, r = lane & 15;
  const uint4* p = reinterpret_cast<const uint4*>(A + (size_t)(row0 + r)*ldA + k0 + half*16);
  FragU f; f.u[0] = p[0]; f.u[1] = p[1];   // K: kbase16..kbase16+15 contiguous
  return f.v;
}
// B-matrix 32x16 fragment where memory is N-major: Mt[n][k] contiguous in k
__device__ __forceinline__ v16bf load_b_frag_nmajor(const unsigned short* Mt, int ldm,
                                                    int n0, int k0, int lane){
  int half = lane >> 4, r = lane & 15;
  const uint4* p = reinterpret_cast<const uint4*>(Mt + (size_t)(n0 + r)*ldm + k0 + half*16);
  FragU f; f.u[0] = p[0]; f.u[1] = p[1];   // K: kbase16..kbase16+15 contiguous
  return f.v;
}
// fragment-major packed weights: 32 lanes x 16 bf16 contiguous
__device__ __forceinline__ v16bf load_b_frag_packed(const unsigned short* Wfrag, int lane){
  const uint4* p = reinterpret_cast<const uint4*>(Wfrag + lane*16);
  FragU f; f.u[0] = p[0]; f.u[1] = p[1];
  return f.v;
}
__device__ __forceinline__ v8f wmma_bf16(v16bf a, v16bf b, v8f c){
  return __builtin_amdgcn_wmma_f32_16x16x32_bf16(false, a, false, b, (short)0, c, false, false);
}

// two K-steps (8KB) of B panel: batch ds loads, then batch WMMAs
__device__ __forceinline__ void gemm_step2(const unsigned short* buf,
                                           v16bf a0, v16bf a1, v8f* acc, int lane){
  v16bf b0 = load_b_frag_packed(buf       , lane);
  v16bf b1 = load_b_frag_packed(buf +  512, lane);
  v16bf b2 = load_b_frag_packed(buf + 1024, lane);
  v16bf b3 = load_b_frag_packed(buf + 1536, lane);
  acc[0] = wmma_bf16(a0, b0, acc[0]);
  acc[1] = wmma_bf16(a0, b1, acc[1]);
  acc[2] = wmma_bf16(a0, b2, acc[2]);
  acc[3] = wmma_bf16(a0, b3, acc[3]);
  v16bf c0 = load_b_frag_packed(buf + 2048, lane);
  v16bf c1 = load_b_frag_packed(buf + 2560, lane);
  v16bf c2 = load_b_frag_packed(buf + 3072, lane);
  v16bf c3 = load_b_frag_packed(buf + 3584, lane);
  acc[0] = wmma_bf16(a1, c0, acc[0]);
  acc[1] = wmma_bf16(a1, c1, acc[1]);
  acc[2] = wmma_bf16(a1, c2, acc[2]);
  acc[3] = wmma_bf16(a1, c3, acc[3]);
}
// async-stage 2 consecutive K-step panels (4KB each) into an 8KB LDS buffer
__device__ __forceinline__ void stage_copy(const unsigned short* Wp, size_t evenPanel,
                                           size_t panelStride, unsigned short* dst, int copyOff){
  async_copy_b128(lds_addr((const char*)dst + copyOff),
                  (const char*)(Wp + evenPanel) + copyOff);
  async_copy_b128(lds_addr((const char*)dst + 4096 + copyOff),
                  (const char*)(Wp + evenPanel + panelStride) + copyOff);
}

// ---------- weight pack: fp32 W[K,N] -> fragment-major bf16 ----------
__global__ void packw_kernel(const float* __restrict__ W, unsigned short* __restrict__ out,
                             int K, int N){
  size_t t = (size_t)blockIdx.x*blockDim.x + threadIdx.x;
  if (t >= (size_t)K*N) return;
  int e    = (int)(t & 15);
  int lane = (int)((t >> 4) & 31);
  size_t rest = t >> 9;                 // kt*nTiles + nt
  int nTiles = N >> 4;
  int kt = (int)(rest / nTiles), nt = (int)(rest % nTiles);
  int half = lane >> 4, r = lane & 15;
  int kk = kt*32 + half*16 + e;         // B-frag: K = kbase16 + e
  int nn = nt*16 + r;
  out[t] = f2bf(W[(size_t)kk*N + nn]);
}

// ---------- layernorm (C=256, one block per row), bf16 out ----------
__global__ void ln_kernel(const float* __restrict__ x, const float* __restrict__ g,
                          const float* __restrict__ b, unsigned short* __restrict__ out,
                          int window_order){
  int rowNat = blockIdx.x;              // b*8192 + h*128 + w
  int c = threadIdx.x;
  __shared__ float red[256];
  float v = x[(size_t)rowNat*C_ + c];
  red[c] = v; __syncthreads();
  for (int s = 128; s > 0; s >>= 1){ if (c < s) red[c] += red[c+s]; __syncthreads(); }
  float mu = red[0] * (1.0f/C_); __syncthreads();
  float d = v - mu;
  red[c] = d*d; __syncthreads();
  for (int s = 128; s > 0; s >>= 1){ if (c < s) red[c] += red[c+s]; __syncthreads(); }
  float var = red[0] * (1.0f/C_);
  float y = d * rsqrtf(var + 1e-5f) * g[c] + b[c];
  size_t dst = rowNat;
  if (window_order){
    int bb = rowNat >> 13, pix = rowNat & 8191, h = pix >> 7, w = pix & 127;
    dst = (size_t)winmap(bb, h, w);
  }
  out[dst*C_ + c] = f2bf(y);
}

// ---------- rotation grid -> nearest-sample index map (H=64,W=128,ws=8) ----------
__global__ void grid_idx_kernel(int* __restrict__ idx){
  int p = blockIdx.x*blockDim.x + threadIdx.x;
  if (p >= 64*128) return;
  int h = p >> 7, w = p & 127;
  int wh = h >> 3, ww = w >> 3, i = h & 7, j = w & 7;
  float seq_i = -1.f + 0.125f + 0.25f*i;
  float seq_j = -1.f + 0.125f + 0.25f*j;
  float t0 = seq_i * 0.125f;            // / nH=8
  float t1 = seq_j * 0.0625f;           // / nW=16
  float beta = (2.f*(wh + 0.5f)*0.125f - 1.f) * (PI_*0.5f);
  float th = t0*(PI_*0.5f) + PI_*0.5f;
  float ph = t1*PI_;
  float sb = sinf(beta), cb = cosf(beta);
  float st = sinf(th),  ct = cosf(th);
  float sp = sinf(ph),  cp = cosf(ph);
  float ca = -sb*st*cp + cb*ct;
  ca = fminf(1.f, fmaxf(-1.f, ca));
  float new_th = acosf(ca);
  float new_ph = atan2f(st*sp, cb*st*cp + sb*ct);
  float lat = (new_th - PI_*0.5f) / (PI_*0.5f);
  lat = fminf(1.f, fmaxf(-1.f, lat));
  float lon = new_ph / PI_;
  float shift = 2.f*(0.5f - (ww + 0.5f)*0.0625f);
  lon -= shift;
  if (lon >  1.f) lon -= 2.f;
  if (lon < -1.f) lon += 2.f;
  int iy = (int)rintf(((lat + 1.f)*64.f  - 1.f)*0.5f);
  int ix = (int)rintf(((lon + 1.f)*128.f - 1.f)*0.5f);
  bool valid = (iy >= 0) && (iy < 64) && (ix >= 0) && (ix < 128);
  int iyc = min(max(iy,0),63), ixc = min(max(ix,0),127);
  idx[p] = valid ? (iyc*128 + ixc) : -1;
}

// ---------- y = grid_sample_nearest(xn), window-token layout ----------
__global__ void gather_y_kernel(const unsigned short* __restrict__ xnw,
                                const int* __restrict__ idx,
                                unsigned short* __restrict__ yw){
  size_t t = (size_t)blockIdx.x*blockDim.x + threadIdx.x;
  if (t >= (size_t)M_*C_) return;
  int c = (int)(t & 255);
  int rowNat = (int)(t >> 8);
  int b = rowNat >> 13, pix = rowNat & 8191, h = pix >> 7, w = pix & 127;
  int mdst = winmap(b, h, w);
  int src = idx[pix];
  unsigned short val = 0;
  if (src >= 0){
    int hs = src >> 7, ws2 = src & 127;
    val = xnw[(size_t)winmap(b, hs, ws2)*C_ + c];
  }
  yw[(size_t)mdst*C_ + c] = val;
}

// ---------- bf16 WMMA GEMM, 128x64 tile, 8 waves ----------
// B staged 2 K-steps at a time, double-buffered in LDS via global_load_async_to_lds_b128.
// mode 0: out bf16 (optional exact GELU); mode 1: f32 residual-add (optional window-reverse remap)
__global__ void gemm_kernel(const unsigned short* __restrict__ A,
                            const unsigned short* __restrict__ Wp,
                            const float* __restrict__ bias,
                            unsigned short* __restrict__ outB,
                            float* __restrict__ outF,
                            const float* __restrict__ resid,
                            int K, int N, int mode, int gelu, int remap){
  __shared__ __align__(16) unsigned short bstage[2][4096];   // 2 x 8KB stages
  int tid = threadIdx.x, wave = tid >> 5, lane = tid & 31;
  int row0  = blockIdx.x*128 + wave*16;
  int ncol0 = blockIdx.y*64;
  int nTiles = N >> 4;
  size_t panel0      = (size_t)(ncol0 >> 4)*512;      // shorts
  size_t panelStride = (size_t)nTiles*512;            // shorts per K-step
  int copyOff = wave*512 + lane*16;                   // bytes: 8 waves x 512B per 4KB panel

  int nstages = K >> 6;                               // 2 K-steps per stage
  stage_copy(Wp, panel0, panelStride, bstage[0], copyOff);

  v8f acc[4] = {};
  v16bf a0 = load_a_frag(A, K, row0, 0,  lane);
  v16bf a1 = load_a_frag(A, K, row0, 32, lane);

  for (int s = 0; s < nstages - 1; ++s){
    int cur = s & 1;
    stage_copy(Wp, panel0 + (size_t)(2*s + 2)*panelStride, panelStride,
               bstage[cur ^ 1], copyOff);
    v16bf n0 = load_a_frag(A, K, row0, (2*s + 2)*32, lane);
    v16bf n1 = load_a_frag(A, K, row0, (2*s + 3)*32, lane);
    __builtin_prefetch(A + (size_t)(row0 + (lane & 15))*K + (2*s + 4)*32, 0, 3);
    asm volatile("s_wait_asynccnt 0x2" ::: "memory");  // current stage's 2 copies done
    __syncthreads();                                   // all waves' portions landed
    gemm_step2(bstage[cur], a0, a1, acc, lane);
    a0 = n0; a1 = n1;
    __syncthreads();                                   // done reading before next overwrite
  }
  // final stage (no further staging)
  asm volatile("s_wait_asynccnt 0x0" ::: "memory");
  __syncthreads();
  gemm_step2(bstage[(nstages - 1) & 1], a0, a1, acc, lane);

  int half = lane >> 4, r = lane & 15;
#pragma unroll
  for (int nt = 0; nt < 4; ++nt){
    int col = ncol0 + nt*16 + r;
    float bv = bias ? bias[col] : 0.f;
#pragma unroll
    for (int vv = 0; vv < 8; ++vv){
      int row = row0 + half*8 + vv;
      float val = acc[nt][vv] + bv;
      if (gelu) val = 0.5f*val*(1.f + erff(val*0.70710678118654752f));
      if (mode == 0){
        outB[(size_t)row*N + col] = f2bf(val);
      } else {
        int nat = row;
        if (remap){                      // window-token -> natural pixel row
          int win = row >> 6, tok = row & 63;
          int b = win >> 7, wh = (win >> 4) & 7, ww = win & 15;
          nat = b*8192 + (wh*8 + (tok >> 3))*128 + (ww*8 + (tok & 7));
        }
        size_t o = (size_t)nat*N + col;
        outF[o] = resid[o] + val;
      }
    }
  }
}

// ---------- attention: one wave (=one head) per 32-thread block ----------
__global__ void attn_kernel(const unsigned short* __restrict__ q,
                            const unsigned short* __restrict__ k,
                            const unsigned short* __restrict__ v,
                            unsigned short* __restrict__ ao){
  __shared__ __align__(16) float          s_lds[64*64];
  __shared__ __align__(16) unsigned short p_lds[64*64];
  __shared__ __align__(16) unsigned short vT[32*64];
  int win = blockIdx.x >> 3, head = blockIdx.x & 7;
  int lane = threadIdx.x & 31;
  int half = lane >> 4, r = lane & 15;
  const float scale = 0.17677669529663687f;     // hd^-0.5, hd=32
  size_t base = (size_t)win*64*C_ + head*32;

  // stage v^T: vT[d][tok] = v[tok][d]
  for (int t = lane; t < 32*64; t += 32){
    int d = t >> 6, tok = t & 63;
    vT[d*64 + tok] = v[base + (size_t)tok*C_ + d];
  }

  // scores: s = (q * scale) @ k^T ; K=32 single WMMA step
  v16bf bk4[4];
#pragma unroll
  for (int nt = 0; nt < 4; ++nt)
    bk4[nt] = load_b_frag_nmajor(k + base, C_, nt*16, 0, lane);
#pragma unroll
  for (int mt = 0; mt < 4; ++mt){
    v16bf a = load_a_frag(q + base, C_, mt*16, 0, lane);
#pragma unroll
    for (int nt = 0; nt < 4; ++nt){
      v8f z = {};
      v8f s = wmma_bf16(a, bk4[nt], z);
#pragma unroll
      for (int vv = 0; vv < 8; ++vv){
        int row = mt*16 + half*8 + vv, col = nt*16 + r;
        s_lds[row*64 + col] = s[vv] * scale;
      }
    }
  }
  __syncthreads();

  // softmax over key dim (64 cols); 2 rows per lane
  for (int rr = lane; rr < 64; rr += 32){
    float mx = -3.0e38f;
    for (int cc = 0; cc < 64; ++cc) mx = fmaxf(mx, s_lds[rr*64 + cc]);
    float sum = 0.f;
    for (int cc = 0; cc < 64; ++cc){
      float e = expf(s_lds[rr*64 + cc] - mx);
      sum += e; s_lds[rr*64 + cc] = e;
    }
    float inv = 1.f / sum;
    for (int cc = 0; cc < 64; ++cc)
      p_lds[rr*64 + cc] = f2bf(s_lds[rr*64 + cc]*inv);
  }
  __syncthreads();

  // out = p @ v : M=64, N=32 (hd), K=64 (2 steps)
  v8f oc[4][2] = {};
#pragma unroll
  for (int kt = 0; kt < 2; ++kt){
    v16bf b0 = load_b_frag_nmajor(vT, 64,  0, kt*32, lane);
    v16bf b1 = load_b_frag_nmajor(vT, 64, 16, kt*32, lane);
#pragma unroll
    for (int mt = 0; mt < 4; ++mt){
      v16bf a = load_a_frag(p_lds, 64, mt*16, kt*32, lane);
      oc[mt][0] = wmma_bf16(a, b0, oc[mt][0]);
      oc[mt][1] = wmma_bf16(a, b1, oc[mt][1]);
    }
  }
#pragma unroll
  for (int mt = 0; mt < 4; ++mt)
#pragma unroll
    for (int nt = 0; nt < 2; ++nt)
#pragma unroll
      for (int vv = 0; vv < 8; ++vv){
        int row = mt*16 + half*8 + vv, d = nt*16 + r;
        ao[base + (size_t)row*C_ + d] = f2bf(oc[mt][nt][vv]);
      }
}

// ---------- lepe: ao += dwconv3x3(q image) (SAME, zero pad), window layout ----------
__global__ void lepe_add_kernel(const unsigned short* __restrict__ q,
                                const float* __restrict__ rw,
                                const float* __restrict__ rb,
                                unsigned short* __restrict__ ao){
  size_t t = (size_t)blockIdx.x*blockDim.x + threadIdx.x;
  if (t >= (size_t)M_*C_) return;
  int c = (int)(t & 255);
  int rowNat = (int)(t >> 8);
  int b = rowNat >> 13, pix = rowNat & 8191, h = pix >> 7, w = pix & 127;
  float acc = rb[c];
#pragma unroll
  for (int dy = -1; dy <= 1; ++dy)
#pragma unroll
    for (int dx = -1; dx <= 1; ++dx){
      int hh = h + dy, ww = w + dx;
      if (hh < 0 || hh >= 64 || ww < 0 || ww >= 128) continue;
      acc += bf2f(q[(size_t)winmap(b, hh, ww)*C_ + c]) * rw[c*9 + (dy+1)*3 + (dx+1)];
    }
  size_t o = (size_t)winmap(b, h, w)*C_ + c;
  ao[o] = f2bf(bf2f(ao[o]) + acc);
}

extern "C" void kernel_launch(void* const* d_in, const int* in_sizes, int n_in,
                              void* d_out, int out_size, void* d_ws, size_t ws_size,
                              hipStream_t stream){
  (void)in_sizes; (void)n_in; (void)out_size; (void)ws_size;
  const float* x     = (const float*)d_in[0];
  const float* n1g   = (const float*)d_in[3];
  const float* n1b   = (const float*)d_in[4];
  const float* wq    = (const float*)d_in[5];
  const float* bq    = (const float*)d_in[6];
  const float* wk    = (const float*)d_in[7];
  const float* bk_   = (const float*)d_in[8];
  const float* wv    = (const float*)d_in[9];
  const float* bv    = (const float*)d_in[10];
  const float* rpew  = (const float*)d_in[11];
  const float* rpeb  = (const float*)d_in[12];
  const float* projw = (const float*)d_in[13];
  const float* projb = (const float*)d_in[14];
  const float* n2g   = (const float*)d_in[15];
  const float* n2b   = (const float*)d_in[16];
  const float* fc1w  = (const float*)d_in[17];
  const float* fc1b  = (const float*)d_in[18];
  const float* fc2w  = (const float*)d_in[19];
  const float* fc2b  = (const float*)d_in[20];

  char* ws = (char*)d_ws;
  const size_t R = (size_t)M_*C_*2;                 // 33,554,432 B per bf16 activation
  unsigned short* xnw = (unsigned short*)(ws);          // xn, later ln2h
  unsigned short* yw  = (unsigned short*)(ws + R);      // y, later ao, later h1 start
  unsigned short* qb  = (unsigned short*)(ws + 2*R);
  unsigned short* kb  = (unsigned short*)(ws + 3*R);
  unsigned short* vb  = (unsigned short*)(ws + 4*R);
  float*          x2  = (float*)(ws + 5*R);            // 67,108,864 B
  char* pw = ws + 5*R + (size_t)M_*C_*4;
  unsigned short* pq    = (unsigned short*)(pw);
  unsigned short* pk    = (unsigned short*)(pw + 131072);
  unsigned short* pv    = (unsigned short*)(pw + 262144);
  unsigned short* pproj = (unsigned short*)(pw + 393216);
  unsigned short* pfc1  = (unsigned short*)(pw + 524288);
  unsigned short* pfc2  = (unsigned short*)(pw + 1048576);
  int*            gidx  = (int*)(pw + 1572864);
  unsigned short* h1   = yw;   // reuses y/ao + q + k + v span (exactly 4R bytes)
  unsigned short* ao   = yw;
  unsigned short* ln2h = xnw;

  // weight packing (fragment-major bf16)
  packw_kernel<<<256,  256, 0, stream>>>(wq,    pq,    C_, C_);
  packw_kernel<<<256,  256, 0, stream>>>(wk,    pk,    C_, C_);
  packw_kernel<<<256,  256, 0, stream>>>(wv,    pv,    C_, C_);
  packw_kernel<<<256,  256, 0, stream>>>(projw, pproj, C_, C_);
  packw_kernel<<<1024, 256, 0, stream>>>(fc1w,  pfc1,  C_, DFF_);
  packw_kernel<<<1024, 256, 0, stream>>>(fc2w,  pfc2,  DFF_, C_);

  // LN1 (window-ordered bf16 out), rotation grid, nearest-gather
  ln_kernel<<<M_, 256, 0, stream>>>(x, n1g, n1b, xnw, 1);
  grid_idx_kernel<<<32, 256, 0, stream>>>(gidx);
  gather_y_kernel<<<M_, 256, 0, stream>>>(xnw, gidx, yw);

  // QKV GEMMs
  gemm_kernel<<<dim3(512,4), 256, 0, stream>>>(xnw, pq, bq,  qb, nullptr, nullptr, C_, C_, 0, 0, 0);
  gemm_kernel<<<dim3(512,4), 256, 0, stream>>>(yw,  pk, bk_, kb, nullptr, nullptr, C_, C_, 0, 0, 0);
  gemm_kernel<<<dim3(512,4), 256, 0, stream>>>(xnw, pv, bv,  vb, nullptr, nullptr, C_, C_, 0, 0, 0);

  // attention (1024 windows x 8 heads), then lepe add
  attn_kernel<<<8192, 32, 0, stream>>>(qb, kb, vb, ao);
  lepe_add_kernel<<<M_, 256, 0, stream>>>(qb, rpew, rpeb, ao);

  // proj + window-reverse + residual -> x2 (f32)
  gemm_kernel<<<dim3(512,4), 256, 0, stream>>>(ao, pproj, projb, nullptr, x2, x, C_, C_, 1, 0, 1);

  // LN2 -> MLP -> final residual into d_out
  ln_kernel<<<M_, 256, 0, stream>>>(x2, n2g, n2b, ln2h, 0);
  gemm_kernel<<<dim3(512,16), 256, 0, stream>>>(ln2h, pfc1, fc1b, h1, nullptr, nullptr, C_, DFF_, 0, 1, 0);
  gemm_kernel<<<dim3(512,4),  256, 0, stream>>>(h1, pfc2, fc2b, nullptr, (float*)d_out, x2, DFF_, C_, 1, 0, 0);
}
